// Cedric_attention_52905407152871
// MI455X (gfx1250) — matrix operand order repaired and checked by reading, not compiled, not run on hardware
//
#include <hip/hip_runtime.h>

// ---------------------------------------------------------------------------
// MI455X (gfx1250, wave32) attention pipeline: bf16 WMMA + TDM tile DMA.
// ---------------------------------------------------------------------------

typedef __bf16 bf16_t;
typedef __attribute__((ext_vector_type(16))) __bf16 v16bf;
typedef __attribute__((ext_vector_type(8)))  __bf16 v8bf;
typedef __attribute__((ext_vector_type(8)))  float  v8f;
typedef __attribute__((ext_vector_type(4)))  unsigned int u32x4;
typedef __attribute__((ext_vector_type(8)))  int i32x8;
typedef __attribute__((ext_vector_type(4)))  int i32x4;

union AFrag { v16bf v; v8bf h[2]; };

#define DIMSZ 2048
#define NH    16
#define NKV   4
#define HDIM  128
#define BATCH 2
#define SEQ   2048
#define MROWS (BATCH * SEQ)   /* 4096 */

#if __has_builtin(__builtin_amdgcn_tensor_load_to_lds) && __has_builtin(__builtin_amdgcn_s_wait_tensorcnt)
#define CED_HAS_TDM 1
#else
#define CED_HAS_TDM 0
#endif

#if CED_HAS_TDM
// 2D tile DMA global -> LDS via Tensor Data Mover (CDNA5 ISA).
// tw/th: tile width/height in elements (bf16); strd: tensor row stride (elements).
// padi: pad_interval code (log2(DWORDs/2)); pada: pad_amount code (DWORDs-1).
static __device__ __forceinline__ void ced_tdm_load_2d(unsigned lds_addr, const bf16_t* g,
                                                       unsigned tw, unsigned th,
                                                       unsigned long long strd,
                                                       int padi, int pada) {
  unsigned long long ga = (unsigned long long)g;
  u32x4 g0 = { 1u,                                   // count=1, user descriptor
               lds_addr,                             // LDS byte address
               (unsigned)ga,                         // global_addr[31:0]
               (unsigned)((ga >> 32) & 0x01ffffffu)  // global_addr[56:32]
                 | 0x80000000u };                    // type=2 ("image")
  i32x8 g1 = { (int)((1u << 16)                      // data_size = 2 bytes
                   | (1u << 20)                      // pad_enable (LDS row pad)
                   | ((unsigned)padi << 22)
                   | ((unsigned)pada << 25)),
               (int)((tw & 0xffffu) << 16),                              // tensor_dim0 lo16 @ [63:48]
               (int)(((tw >> 16) & 0xffffu) | ((th & 0xffffu) << 16)),   // dim0 hi | dim1 lo
               (int)(((th >> 16) & 0xffffu) | ((tw & 0xffffu) << 16)),   // dim1 hi | tile_dim0
               (int)(th & 0xffffu),                                      // tile_dim1 (tile_dim2=0)
               (int)(strd & 0xffffffffu),                                // dim0_stride lo32
               (int)((strd >> 32) & 0xffffu),                            // dim0_stride hi16
               0 };
  i32x4 gz = { 0, 0, 0, 0 };
#if defined(__clang_major__) && __clang_major__ >= 23
  i32x8 gz8 = { 0, 0, 0, 0, 0, 0, 0, 0 };
  __builtin_amdgcn_tensor_load_to_lds(g0, g1, gz, gz, gz8, 0);
#else
  __builtin_amdgcn_tensor_load_to_lds(g0, g1, gz, gz, 0);
#endif
}
// LDS aperture: flat addr[31:0] is the LDS byte address (ISA §10.2).
static __device__ __forceinline__ unsigned ced_lds_off(const void* p) {
  return (unsigned)(unsigned long long)p;
}
#endif

// ============================= conversion ==================================
__global__ void ced_conv_bf16(const float* __restrict__ X, bf16_t* __restrict__ Y, int n) {
  int i = blockIdx.x * blockDim.x + threadIdx.x;
  if (i < n) Y[i] = (bf16_t)X[i];
}

// W [K x N] f32 row-major  ->  Wt [N x K] bf16 row-major
__global__ void ced_transpose_w(const float* __restrict__ W, bf16_t* __restrict__ Wt,
                                int K, int N) {
  int idx = blockIdx.x * blockDim.x + threadIdx.x;
  if (idx >= K * N) return;
  int k = idx % K, n = idx / K;
  Wt[(size_t)n * K + k] = (bf16_t)W[(size_t)k * N + n];
}

// ============================= generic GEMM ================================
// C[M x N] f32 = A[M x K] bf16 * Bt[N x K]^T bf16 via v_wmma_f32_16x16x32_bf16.
// Block: 256 threads (8 waves). Block tile 128x128, wave tile 32x64, K step 32.
// Tiles arrive by double-buffered TDM DMA (wave 0 issues, TENSORcnt-gated).
#define GLDA 40   // padded LDS row stride (elements); 80B keeps 16B alignment

__global__ __launch_bounds__(256) void ced_gemm_bf16(const bf16_t* __restrict__ A,
                                                     const bf16_t* __restrict__ Bt,
                                                     float* __restrict__ C,
                                                     int M, int N, int K) {
  __shared__ bf16_t Alds[2][128 * GLDA];
  __shared__ bf16_t Blds[2][128 * GLDA];

  const int tid  = threadIdx.x;
  const int lane = tid & 31;
  const int w    = tid >> 5;        // 8 waves
  const int lr   = lane & 15;
  const int lg   = lane >> 4;
  const int wm   = w & 3;           // M sub-block 0..3 (32 rows each)
  const int wn   = w >> 2;          // N sub-block 0..1 (64 cols each)

  const int m0 = blockIdx.y * 128;
  const int n0 = blockIdx.x * 128;
  const int nsteps = K >> 5;

  v8f acc[2][4] = {};
  int cur = 0;

#if CED_HAS_TDM
  if (w == 0) {   // prologue: stage 0 tiles   (pad: 16DW interval -> code3, 4DW -> code3)
    ced_tdm_load_2d(ced_lds_off(&Alds[0][0]), &A[(size_t)m0 * K], 32, 128, (unsigned long long)K, 3, 3);
    ced_tdm_load_2d(ced_lds_off(&Blds[0][0]), &Bt[(size_t)n0 * K], 32, 128, (unsigned long long)K, 3, 3);
  }
#endif

  for (int i = 0; i < nsteps; ++i) {
#if CED_HAS_TDM
    if (w == 0) {
      if (i + 1 < nsteps) {
        int k1 = (i + 1) << 5;
        ced_tdm_load_2d(ced_lds_off(&Alds[cur ^ 1][0]), &A[(size_t)m0 * K + k1], 32, 128,
                        (unsigned long long)K, 3, 3);
        ced_tdm_load_2d(ced_lds_off(&Blds[cur ^ 1][0]), &Bt[(size_t)n0 * K + k1], 32, 128,
                        (unsigned long long)K, 3, 3);
        __builtin_amdgcn_s_wait_tensorcnt(2);   // stage i's 2 DMAs done, next 2 in flight
      } else {
        __builtin_amdgcn_s_wait_tensorcnt(0);
      }
    }
    __syncthreads();
#else
    __syncthreads();
    {
      int k0 = i << 5;
      #pragma unroll
      for (int it = 0; it < 2; ++it) {
        int ch = tid + it * 256;                 // 512 chunks of 16B per tile
        int r = ch >> 2, c = (ch & 3) * 8;
        *(v8bf*)&Alds[cur][r * GLDA + c] = *(const v8bf*)&A[(size_t)(m0 + r) * K + k0 + c];
        *(v8bf*)&Blds[cur][r * GLDA + c] = *(const v8bf*)&Bt[(size_t)(n0 + r) * K + k0 + c];
      }
    }
    __syncthreads();
#endif

    AFrag a0, a1;
    {
      const bf16_t* ap = &Alds[cur][(wm * 32 + lr) * GLDA + lg * 8];
      a0.h[0] = *(const v8bf*)ap;
      a0.h[1] = *(const v8bf*)(ap + 16);
      ap += 16 * GLDA;
      a1.h[0] = *(const v8bf*)ap;
      a1.h[1] = *(const v8bf*)(ap + 16);
    }
    #pragma unroll
    for (int nt = 0; nt < 4; ++nt) {
      AFrag b;
      const bf16_t* bp = &Blds[cur][(wn * 64 + nt * 16 + lr) * GLDA + lg * 8];
      b.h[0] = *(const v8bf*)bp;
      b.h[1] = *(const v8bf*)(bp + 16);
      acc[0][nt] = __builtin_amdgcn_wmma_f32_16x16x32_bf16(
          false, a0.v, false, b.v, (short)0, acc[0][nt], false, false);
      acc[1][nt] = __builtin_amdgcn_wmma_f32_16x16x32_bf16(
          false, a1.v, false, b.v, (short)0, acc[1][nt], false, false);
    }
    __syncthreads();
    cur ^= 1;
  }

  #pragma unroll
  for (int mi = 0; mi < 2; ++mi)
    #pragma unroll
    for (int nt = 0; nt < 4; ++nt)
      #pragma unroll
      for (int j = 0; j < 8; ++j) {
        int row = m0 + wm * 32 + mi * 16 + j + 8 * lg;   // WMMA C layout
        int col = n0 + wn * 64 + nt * 16 + lr;
        C[(size_t)row * N + col] = acc[mi][nt][j];
      }
}

// ============================= RoPE + layout ===============================
// Reference quirk: apply_rope(x, sin_t, cos_t) swaps the tables:
//   re' = re*sin - im*cos ; im' = re*cos + im*sin
__global__ void ced_rope_convert(const float* __restrict__ X, bf16_t* __restrict__ Y,
                                 int S, int Hn, int do_rope) {
  int idx = blockIdx.x * blockDim.x + threadIdx.x;
  int total = BATCH * S * Hn * (HDIM / 2);
  if (idx >= total) return;
  int p = idx & 63; int t = idx >> 6;
  int h = t % Hn;   t /= Hn;
  int s = t % S;    int b = t / S;
  size_t src = ((size_t)(b * S + s) * Hn + h) * HDIM + 2 * p;
  float re = X[src], im = X[src + 1];
  float o0 = re, o1 = im;
  if (do_rope) {
    float freq = __powf(10000.0f, -(float)(2 * p) * (1.0f / HDIM));
    float ang = (float)s * freq;
    float sn, cs;
    __sincosf(ang, &sn, &cs);
    o0 = re * sn - im * cs;
    o1 = re * cs + im * sn;
  }
  size_t dst = (((size_t)(b * Hn + h) * S + s) * HDIM) + 2 * p;
  Y[dst] = (bf16_t)o0;
  Y[dst + 1] = (bf16_t)o1;
}

// V: [B*S][NKV*128] f32 -> Vt: [B][NKV][128][S] bf16 (d-major = WMMA-B K-major)
__global__ void ced_v_transpose(const float* __restrict__ X, bf16_t* __restrict__ Y, int S) {
  int idx = blockIdx.x * blockDim.x + threadIdx.x;
  int total = BATCH * S * NKV * HDIM;
  if (idx >= total) return;
  int d = idx & (HDIM - 1); int t = idx >> 7;
  int h = t % NKV; t /= NKV;
  int s = t % S;   int b = t / S;
  float v = X[((size_t)(b * S + s) * NKV + h) * HDIM + d];
  Y[(((size_t)(b * NKV + h) * HDIM + d) * S) + s] = (bf16_t)v;
}

// ============================= flash attention =============================
// Per block: (batch b, head h, 64-row q tile). 128 threads = 4 waves;
// wave w owns q rows [q0+16w, +16). Causal: kv tiles kt <= qt only.
#define QLD 136   // 128 + 8 pad  (pad codes: 64DW interval -> 5, 4DW -> 3)
#define VLD 72    //  64 + 8 pad  (pad codes: 32DW interval -> 4, 4DW -> 3)

__global__ __launch_bounds__(128) void ced_flash_attn(const bf16_t* __restrict__ Q,
                                                      const bf16_t* __restrict__ K,
                                                      const bf16_t* __restrict__ Vt,
                                                      bf16_t* __restrict__ O) {
  __shared__ bf16_t Qs[64 * QLD];
  __shared__ bf16_t Ks[64 * QLD];
  __shared__ bf16_t Vts[128 * VLD];
  __shared__ bf16_t Ps[64 * VLD];

  const int tid  = threadIdx.x;
  const int lane = tid & 31;
  const int w    = tid >> 5;
  const int lr   = lane & 15;
  const int lg   = lane >> 4;

  const int qt  = blockIdx.x;
  const int h   = blockIdx.y;
  const int b   = blockIdx.z;
  const int kvh = h >> 2;              // n_rep = 4 (jnp.repeat semantics)
  const int q0  = qt * 64;

  const bf16_t* Qbase = Q  + ((size_t)(b * NH  + h)   * SEQ + q0) * HDIM;
  const bf16_t* Kbase = K  + ((size_t)(b * NKV + kvh) * SEQ) * HDIM;
  const bf16_t* Vbase = Vt + ((size_t)(b * NKV + kvh) * HDIM) * SEQ;

  // Q tile: 64 x 128 = 1024 16B chunks, cooperative (loaded once)
  #pragma unroll
  for (int it = 0; it < 8; ++it) {
    int ch = tid + it * 128;
    int r = ch >> 4, c = (ch & 15) * 8;
    *(v8bf*)&Qs[r * QLD + c] = *(const v8bf*)&Qbase[(size_t)r * HDIM + c];
  }

  v8f o[8] = {};
  float mj[8], lj[8];
  #pragma unroll
  for (int j = 0; j < 8; ++j) { mj[j] = -1e30f; lj[j] = 0.0f; }

  const float scale = 0.08838834764831845f;   // 1/sqrt(128)

  for (int kt = 0; kt <= qt; ++kt) {
    const int kv0 = kt * 64;
    const bool diag = (kt == qt);
    __syncthreads();                         // previous tiles fully consumed
#if CED_HAS_TDM
    if (w == 0) {
      ced_tdm_load_2d(ced_lds_off(&Ks[0]),  Kbase + (size_t)kv0 * HDIM, 128, 64,
                      (unsigned long long)HDIM, 5, 3);
      ced_tdm_load_2d(ced_lds_off(&Vts[0]), Vbase + kv0, 64, 128,
                      (unsigned long long)SEQ, 4, 3);
      __builtin_amdgcn_s_wait_tensorcnt(0);
    }
#else
    #pragma unroll
    for (int it = 0; it < 8; ++it) {
      int ch = tid + it * 128;
      int r = ch >> 4, c = (ch & 15) * 8;     // K tile 64x128
      *(v8bf*)&Ks[r * QLD + c] = *(const v8bf*)&Kbase[(size_t)(kv0 + r) * HDIM + c];
      int vr = ch >> 3, vc = (ch & 7) * 8;    // Vt tile 128x64
      *(v8bf*)&Vts[vr * VLD + vc] = *(const v8bf*)&Vbase[(size_t)vr * SEQ + kv0 + vc];
    }
#endif
    __syncthreads();

    // ---- S = Q . K^T  (16 WMMAs) ----
    v8f sacc[4] = {};
    #pragma unroll
    for (int kk = 0; kk < 4; ++kk) {
      AFrag a;
      const bf16_t* ap = &Qs[(w * 16 + lr) * QLD + kk * 32 + lg * 8];
      a.h[0] = *(const v8bf*)ap;
      a.h[1] = *(const v8bf*)(ap + 16);
      #pragma unroll
      for (int nt = 0; nt < 4; ++nt) {
        AFrag bb;
        const bf16_t* bp = &Ks[(nt * 16 + lr) * QLD + kk * 32 + lg * 8];
        bb.h[0] = *(const v8bf*)bp;
        bb.h[1] = *(const v8bf*)(bp + 16);
        sacc[nt] = __builtin_amdgcn_wmma_f32_16x16x32_bf16(
            false, a.v, false, bb.v, (short)0, sacc[nt], false, false);
      }
    }

    // ---- online softmax (row = j + 8*lg across 16-lane half-groups) ----
    #pragma unroll
    for (int j = 0; j < 8; ++j) {
      const int qrow = q0 + w * 16 + j + 8 * lg;
      float vv[4];
      #pragma unroll
      for (int nt = 0; nt < 4; ++nt) {
        float v = sacc[nt][j] * scale;
        if (diag && (kv0 + nt * 16 + lr > qrow)) v = -1e30f;   // causal mask
        vv[nt] = v;
      }
      float mx = fmaxf(fmaxf(vv[0], vv[1]), fmaxf(vv[2], vv[3]));
      #pragma unroll
      for (int msk = 1; msk < 16; msk <<= 1) mx = fmaxf(mx, __shfl_xor(mx, msk, 32));
      float mnew  = fmaxf(mj[j], mx);
      float alpha = __expf(mj[j] - mnew);
      float rs = 0.0f;
      #pragma unroll
      for (int nt = 0; nt < 4; ++nt) {
        float p = __expf(vv[nt] - mnew);
        rs += p;
        Ps[(w * 16 + j + 8 * lg) * VLD + nt * 16 + lr] = (bf16_t)p;
      }
      #pragma unroll
      for (int msk = 1; msk < 16; msk <<= 1) rs += __shfl_xor(rs, msk, 32);
      lj[j] = lj[j] * alpha + rs;
      mj[j] = mnew;
      #pragma unroll
      for (int dt = 0; dt < 8; ++dt) o[dt][j] *= alpha;
    }

    // ---- O += P . V  (16 WMMAs) ----
    #pragma unroll
    for (int kk2 = 0; kk2 < 64; kk2 += 32) {
      AFrag a;
      const bf16_t* ap = &Ps[(w * 16 + lr) * VLD + kk2 + lg * 8];
      a.h[0] = *(const v8bf*)ap;
      a.h[1] = *(const v8bf*)(ap + 16);
      #pragma unroll
      for (int dt = 0; dt < 8; ++dt) {
        AFrag bb;
        const bf16_t* bp = &Vts[(dt * 16 + lr) * VLD + kk2 + lg * 8];
        bb.h[0] = *(const v8bf*)bp;
        bb.h[1] = *(const v8bf*)(bp + 16);
        o[dt] = __builtin_amdgcn_wmma_f32_16x16x32_bf16(
            false, a.v, false, bb.v, (short)0, o[dt], false, false);
      }
    }
  }

  // ---- epilogue: normalize, store bf16 in [B*S][2048] for the w0 GEMM ----
  #pragma unroll
  for (int dt = 0; dt < 8; ++dt)
    #pragma unroll
    for (int j = 0; j < 8; ++j) {
      float val = o[dt][j] / lj[j];
      size_t row = (size_t)b * SEQ + q0 + w * 16 + j + 8 * lg;
      O[row * DIMSZ + h * HDIM + dt * 16 + lr] = (bf16_t)val;
    }
}

// ============================= launch ======================================
extern "C" void kernel_launch(void* const* d_in, const int* in_sizes, int n_in,
                              void* d_out, int out_size, void* d_ws, size_t ws_size,
                              hipStream_t stream) {
  (void)in_sizes; (void)n_in; (void)out_size; (void)ws_size;
  const float* x  = (const float*)d_in[0];
  const float* wq = (const float*)d_in[1];
  const float* wk = (const float*)d_in[2];
  const float* wv = (const float*)d_in[3];
  const float* w0 = (const float*)d_in[4];

  size_t off = 0;
  auto carve = [&](size_t bytes) {
    void* p = (char*)d_ws + off;
    off += (bytes + 255) & ~(size_t)255;
    return p;
  };
  bf16_t* xbf = (bf16_t*)carve((size_t)MROWS * DIMSZ * 2);
  bf16_t* wqt = (bf16_t*)carve((size_t)DIMSZ * DIMSZ * 2);
  bf16_t* wkt = (bf16_t*)carve((size_t)512 * DIMSZ * 2);
  bf16_t* wvt = (bf16_t*)carve((size_t)512 * DIMSZ * 2);
  bf16_t* w0t = (bf16_t*)carve((size_t)DIMSZ * DIMSZ * 2);
  float*  qf  = (float*) carve((size_t)MROWS * DIMSZ * 4);
  float*  kf  = (float*) carve((size_t)MROWS * 512 * 4);
  float*  vf  = (float*) carve((size_t)MROWS * 512 * 4);
  bf16_t* qb  = (bf16_t*)carve((size_t)MROWS * DIMSZ * 2);   // Q [B,H,S,HD]
  bf16_t* kb  = (bf16_t*)carve((size_t)MROWS * 512 * 2);     // K [B,KVH,S,HD]
  bf16_t* vtb = (bf16_t*)carve((size_t)MROWS * 512 * 2);     // Vt [B,KVH,HD,S]
  bf16_t* ob  = (bf16_t*)carve((size_t)MROWS * DIMSZ * 2);   // attn out bf16

  {
    int n = MROWS * DIMSZ;
    ced_conv_bf16<<<(n + 255) / 256, 256, 0, stream>>>(x, xbf, n);
  }
  ced_transpose_w<<<(DIMSZ * DIMSZ + 255) / 256, 256, 0, stream>>>(wq, wqt, DIMSZ, DIMSZ);
  ced_transpose_w<<<(DIMSZ * 512  + 255) / 256, 256, 0, stream>>>(wk, wkt, DIMSZ, 512);
  ced_transpose_w<<<(DIMSZ * 512  + 255) / 256, 256, 0, stream>>>(wv, wvt, DIMSZ, 512);
  ced_transpose_w<<<(DIMSZ * DIMSZ + 255) / 256, 256, 0, stream>>>(w0, w0t, DIMSZ, DIMSZ);

  ced_gemm_bf16<<<dim3(DIMSZ / 128, MROWS / 128), 256, 0, stream>>>(xbf, wqt, qf, MROWS, DIMSZ, DIMSZ);
  ced_gemm_bf16<<<dim3(512  / 128, MROWS / 128), 256, 0, stream>>>(xbf, wkt, kf, MROWS, 512, DIMSZ);
  ced_gemm_bf16<<<dim3(512  / 128, MROWS / 128), 256, 0, stream>>>(xbf, wvt, vf, MROWS, 512, DIMSZ);

  {
    int nq = BATCH * SEQ * NH * (HDIM / 2);
    ced_rope_convert<<<(nq + 255) / 256, 256, 0, stream>>>(qf, qb, SEQ, NH, 1);
    int nk = BATCH * SEQ * NKV * (HDIM / 2);
    ced_rope_convert<<<(nk + 255) / 256, 256, 0, stream>>>(kf, kb, SEQ, NKV, 1);
    int nv = BATCH * SEQ * NKV * HDIM;
    ced_v_transpose<<<(nv + 255) / 256, 256, 0, stream>>>(vf, vtb, SEQ);
  }

  ced_flash_attn<<<dim3(SEQ / 64, NH, BATCH), 128, 0, stream>>>(qb, kb, vtb, ob);

  ced_gemm_bf16<<<dim3(DIMSZ / 128, MROWS / 128), 256, 0, stream>>>(ob, w0t, (float*)d_out,
                                                                    MROWS, DIMSZ, DIMSZ);
}
